// BraggNN_15779709845809
// MI455X (gfx1250) — matrix operand order repaired and checked by reading, not compiled, not run on hardware
//
#include <hip/hip_runtime.h>

typedef __attribute__((ext_vector_type(16))) _Float16 v16h;
typedef __attribute__((ext_vector_type(8)))  _Float16 v8h;
typedef __attribute__((ext_vector_type(8)))  float    v8f;

#define LEAKY(v) ((v) > 0.0f ? (v) : 0.01f * (v))

struct Params {
    const float *x;
    const float *c1w, *c1b;
    const float *thw, *thb, *phw, *phb, *gw, *gb, *ow, *ob;
    const float *c2w, *c2b, *c3w, *c3b;
    const float *f1w, *f1b, *f2w, *f2b, *f3w, *f3b, *f4w, *f4b, *f5w, *f5b;
    float *out;
};

// f16 arena in d_ws (offsets in halves, all 32B-aligned).
// All B matrices stored TRANSPOSED [N][K] -> one 32B load per B fragment.
#define W1T_OFF  0       // conv1  [64 oc][32 k]       (k>=9 zero)
#define PROJ_OFF 2048    // th/ph/g [3][32 i][64 c]
#define OWT_OFF  8192    // o_w    [64 c][32 i]
#define W2T_OFF  10240   // conv2  [32 oc][576 k], k=(ky*3+kx)*64+c
#define W3T_OFF  28672   // conv3  [16 oc][288 k]      (oc>=8 zero)
#define F1T_OFF  33280   // fc1    [64 j][224 k]       (k>=200 zero)
#define F2T_OFF  47616   // fc2    [32 j][64 k]
#define F3T_OFF  49664   // fc3    [16 j][32 k]
#define F4T_OFF  50176   // fc4    [16 j][32 k]        (j>=8 | k>=16 zero)
#define F5T_OFF  50688   // fc5    [16 j][32 k]        (j>=2 | k>=8 zero)
#define F_OFF    51200   // features [B][224] f16      (k>=200 zero)
// d_ws bytes needed: (51200 + B*224)*2  ~= 7.2 MB for B=16384

// ---- wave32 WMMA fragment access (ISA 7.12.2) ------------------------------

static __device__ __forceinline__ v8f wmma(const v16h& a, const v16h& b, const v8f& c) {
    return __builtin_amdgcn_wmma_f32_16x16x32_f16(false, a, false, b, (short)0, c,
                                                  false, false);
}

// A fragment from row-major [M][K]: rowbase = &src[m*ld] (m = lane&15).
static __device__ __forceinline__ v16h load_a16(const _Float16* rowbase, int lane) {
    const int kb = (lane & 16) ? 16 : 0;
    return *(const v16h*)(rowbase + kb);
}

// B fragment from TRANSPOSED [N][K]: colbase = &srcT[n*ld] (n = lane&15).
static __device__ __forceinline__ v16h load_b16(const _Float16* colbase, int lane) {
    const int kb = (lane & 16) ? 16 : 0;
    return *(const v16h*)(colbase + kb);
}

// C/D (16x16 f32): lane holds col n=lane&15, reg r holds row m = r + (lane&16?8:0).
template <typename F>
static __device__ __forceinline__ void store_c(int lane, const v8f& c, F&& put) {
    const int mb = (lane & 16) ? 8 : 0;
    const int n  = lane & 15;
#pragma unroll
    for (int r = 0; r < 8; ++r) put(mb + r, n, c[r]);
}

// ---- prep kernel: convert + transpose ALL weights to f16 in d_ws -----------

__global__ void prep_weights(Params p, _Float16* ws) {
    const int stride = gridDim.x * blockDim.x;
    const int tid0   = blockIdx.x * blockDim.x + threadIdx.x;

    for (int i = tid0; i < 2048; i += stride) {               // w1t[oc][k]
        const int oc = i >> 5, k = i & 31;
        ws[W1T_OFF + i] = (k < 9) ? (_Float16)p.c1w[oc * 9 + k] : (_Float16)0;
    }
    for (int i = tid0; i < 6144; i += stride) {               // projT[which][i][c]
        const int which = i >> 11, r = i & 2047, ii = r >> 6, c = r & 63;
        const float* W = (which == 0) ? p.thw : (which == 1) ? p.phw : p.gw;
        ws[PROJ_OFF + i] = (_Float16)W[c * 32 + ii];
    }
    for (int i = tid0; i < 2048; i += stride) {               // owt[c][i]
        const int c = i >> 5, ii = i & 31;
        ws[OWT_OFF + i] = (_Float16)p.ow[ii * 64 + c];
    }
    for (int i = tid0; i < 18432; i += stride) {              // w2t[oc][k]
        const int oc = i / 576, k = i - oc * 576;
        const int kyx = k >> 6, c = k & 63;
        ws[W2T_OFF + i] = (_Float16)p.c2w[(oc * 64 + c) * 9 + kyx];
    }
    for (int i = tid0; i < 4608; i += stride) {               // w3t[oc][k]
        const int oc = i / 288, k = i - oc * 288;
        const int kyx = k >> 5, c = k & 31;
        ws[W3T_OFF + i] =
            (oc < 8) ? (_Float16)p.c3w[(oc * 32 + c) * 9 + kyx] : (_Float16)0;
    }
    for (int i = tid0; i < 14336; i += stride) {              // f1t[j][k]
        const int j = i / 224, k = i - j * 224;
        ws[F1T_OFF + i] = (k < 200) ? (_Float16)p.f1w[k * 64 + j] : (_Float16)0;
    }
    for (int i = tid0; i < 2048; i += stride) {               // f2t[j][k]
        const int j = i >> 6, k = i & 63;
        ws[F2T_OFF + i] = (_Float16)p.f2w[k * 32 + j];
    }
    for (int i = tid0; i < 512; i += stride) {                // f3t[j][k]
        const int j = i >> 5, k = i & 31;
        ws[F3T_OFF + i] = (_Float16)p.f3w[k * 16 + j];
    }
    for (int i = tid0; i < 512; i += stride) {                // f4t[j][k]
        const int j = i >> 5, k = i & 31;
        ws[F4T_OFF + i] =
            (j < 8 && k < 16) ? (_Float16)p.f4w[k * 8 + j] : (_Float16)0;
    }
    for (int i = tid0; i < 512; i += stride) {                // f5t[j][k]
        const int j = i >> 5, k = i & 31;
        ws[F5T_OFF + i] =
            (j < 2 && k < 8) ? (_Float16)p.f5w[k * 2 + j] : (_Float16)0;
    }
}

// ---- main fused kernel: conv1 + NLB attention + conv2 + conv3 --------------
// One sample per workgroup (4 wave32). Features written f16 to ws[F_OFF].

__global__ __launch_bounds__(128) void braggnn_fused(Params p, _Float16* ws) {
    const int s    = blockIdx.x;
    const int tid  = threadIdx.x;
    const int wave = tid >> 5;
    const int lane = tid & 31;
    const int m    = lane & 15;

    __shared__ __align__(32) _Float16 x_s[160];        // normalized input, zero tail
    __shared__ __align__(32) _Float16 h1_s[96 * 64];   // [pos][ch], rows>=81 zero
    __shared__ __align__(32) _Float16 t_s[96 * 32];    // im2col A1 / theta / y
    __shared__ __align__(32) _Float16 p_s[96 * 32];    // P[pos][i] (phi^T, B-ready)
    __shared__ __align__(32) _Float16 g_t[32 * 96];    // g^T [i][pos] (B-ready)
    __shared__ __align__(32) _Float16 attn_s[96 * 96]; // scores -> softmax
    __shared__ __align__(32) _Float16 h2_s[64 * 32];   // [pos7x7][ch], rows>=49 zero
    __shared__ float mm_s[128];
    __shared__ float red_s[2];
    __shared__ float bias_s[264];  // c1b@0 thb@64 phb@96 gb@128 ob@160 c2b@224 c3b@256

    // ---- phase 0: biases + per-sample min-max normalization ----------------
    {
        for (int i = tid; i < 264; i += 128) {
            float v;
            if (i < 64)       v = p.c1b[i];
            else if (i < 96)  v = p.thb[i - 64];
            else if (i < 128) v = p.phb[i - 96];
            else if (i < 160) v = p.gb[i - 128];
            else if (i < 224) v = p.ob[i - 160];
            else if (i < 256) v = p.c2b[i - 224];
            else              v = p.c3b[i - 256];
            bias_s[i] = v;
        }
        const float* xs = p.x + (size_t)s * 121;
        const float  xv = (tid < 121) ? xs[tid] : xs[0];
        mm_s[tid] = xv;
        __syncthreads();
        if (tid == 0) {
            float mn = mm_s[0], mx = mm_s[0];
            for (int i = 1; i < 121; ++i) {
                mn = fminf(mn, mm_s[i]);
                mx = fmaxf(mx, mm_s[i]);
            }
            red_s[0] = mn;
            red_s[1] = 1.0f / (mx - mn);
        }
        __syncthreads();
        x_s[tid] = (tid < 121) ? (_Float16)((xv - red_s[0]) * red_s[1]) : (_Float16)0;
        if (tid < 32) x_s[128 + tid] = (_Float16)0;
        __syncthreads();
    }

    // ---- phase 0.5: stage conv1 im2col A [96 pos][32 k] into t_s (once) ----
    for (int i = tid; i < 3072; i += 128) {
        const int pos = i >> 5, k = i & 31;
        const int py = pos / 9, px = pos - py * 9;
        const int ky = k / 3, kx = k - ky * 3;
        int idx = (py + ky) * 11 + px + kx;
        idx = (pos < 81 && k < 9) ? idx : 159;  // branchless: read zero tail
        t_s[i] = x_s[idx];
    }
    __syncthreads();

    // ---- phase 1: conv1 (1->64); GEMM [96x32pad]@[32x64], all vector loads -
    for (int job = wave; job < 24; job += 4) {
        const int mt = job >> 2, nt = job & 3;
        v8f acc = wmma(load_a16(t_s + (mt * 16 + m) * 32, lane),
                       load_b16(ws + W1T_OFF + (nt * 16 + m) * 32, lane), v8f{});
        store_c(lane, acc, [&](int mm, int n, float v) {
            const int pos2 = mt * 16 + mm, oc = nt * 16 + n;
            const float r = LEAKY(v + bias_s[oc]);
            h1_s[pos2 * 64 + oc] = (pos2 < 81) ? (_Float16)r : (_Float16)0;
        });
    }
    __syncthreads();

    // ---- phase 2: theta / phi / g projections [96x64]@[64x32] --------------
    for (int job = wave; job < 36; job += 4) {
        const int which = job / 12, t = job - which * 12;
        const int mt = t >> 1, nt = t & 1;
        const _Float16* Arow = h1_s + (mt * 16 + m) * 64;
        const _Float16* Bcol = ws + PROJ_OFF + which * 2048 + (nt * 16 + m) * 64;
        v8f acc = {};
#pragma unroll
        for (int kc = 0; kc < 2; ++kc)
            acc = wmma(load_a16(Arow + kc * 32, lane), load_b16(Bcol + kc * 32, lane), acc);
        const float* Bv = bias_s + 64 + which * 32;
        if (which != 2) {
            _Float16* D = (which == 0) ? t_s : p_s;
            store_c(lane, acc, [&](int mm, int n, float v) {
                D[(mt * 16 + mm) * 32 + nt * 16 + n] = (_Float16)(v + Bv[nt * 16 + n]);
            });
        } else {  // g stored transposed [i][pos] for phase-4 B side
            store_c(lane, acc, [&](int mm, int n, float v) {
                g_t[(nt * 16 + n) * 96 + mt * 16 + mm] = (_Float16)(v + Bv[nt * 16 + n]);
            });
        }
    }
    __syncthreads();

    // ---- phase 3: scores = theta @ phi ([96x32]@[32x96], B = p_s[N][K]) ----
    for (int job = wave; job < 36; job += 4) {
        const int mt = job / 6, nt = job - mt * 6;
        v8f acc = wmma(load_a16(t_s + (mt * 16 + m) * 32, lane),
                       load_b16(p_s + (nt * 16 + m) * 32, lane), v8f{});
        store_c(lane, acc, [&](int mm, int n, float v) {
            attn_s[(mt * 16 + mm) * 96 + nt * 16 + n] = (_Float16)v;
        });
    }
    __syncthreads();

    // ---- softmax: vectorized row loads, all guards const-folded ------------
    if (tid < 96) {
        _Float16* row = attn_s + tid * 96;
        v8h ch[12];
#pragma unroll
        for (int c8 = 0; c8 < 12; ++c8) ch[c8] = *(const v8h*)(row + c8 * 8);
        float mx = -1e30f;
#pragma unroll
        for (int c8 = 0; c8 < 12; ++c8)
#pragma unroll
            for (int j = 0; j < 8; ++j)
                if (c8 * 8 + j < 81) mx = fmaxf(mx, (float)ch[c8][j]);
        float sum = 0.0f;
#pragma unroll
        for (int c8 = 0; c8 < 12; ++c8) {
            v8h o;
#pragma unroll
            for (int j = 0; j < 8; ++j) {
                const float e =
                    (c8 * 8 + j < 81) ? __expf((float)ch[c8][j] - mx) : 0.0f;
                sum += e;
                o[j] = (_Float16)e;
            }
            ch[c8] = o;
        }
        const float r = 1.0f / sum;
#pragma unroll
        for (int c8 = 0; c8 < 12; ++c8) {
            v8h o;
#pragma unroll
            for (int j = 0; j < 8; ++j) o[j] = (_Float16)((float)ch[c8][j] * r);
            *(v8h*)(row + c8 * 8) = o;
        }
    }
    __syncthreads();

    // ---- phase 4: y = attn @ g ([96x96]@[96x32], B = g_t[N][K]) -> t_s -----
    for (int job = wave; job < 12; job += 4) {
        const int mt = job >> 1, nt = job & 1;
        const _Float16* Arow = attn_s + (mt * 16 + m) * 96;
        const _Float16* Bcol = g_t + (nt * 16 + m) * 96;
        v8f acc = {};
#pragma unroll
        for (int kc = 0; kc < 3; ++kc)
            acc = wmma(load_a16(Arow + kc * 32, lane), load_b16(Bcol + kc * 32, lane), acc);
        store_c(lane, acc, [&](int mm, int n, float v) {
            t_s[(mt * 16 + mm) * 32 + nt * 16 + n] = (_Float16)v;
        });
    }
    __syncthreads();

    // ---- phase 5: out = y @ o_w + o_b + h1 (residual into h1_s) ------------
    for (int job = wave; job < 24; job += 4) {
        const int mt = job >> 2, nt = job & 3;
        v8f acc = wmma(load_a16(t_s + (mt * 16 + m) * 32, lane),
                       load_b16(ws + OWT_OFF + (nt * 16 + m) * 32, lane), v8f{});
        store_c(lane, acc, [&](int mm, int n, float v) {
            const int pos = mt * 16 + mm, c = nt * 16 + n;
            const float r = v + bias_s[160 + c] + (float)h1_s[pos * 64 + c];
            h1_s[pos * 64 + c] = (pos < 81) ? (_Float16)r : (_Float16)0;
        });
    }
    __syncthreads();

    // ---- phase 6: conv2 (64->32, 9x9->7x7) implicit GEMM, K=576 ------------
    for (int job = wave; job < 8; job += 4) {
        const int mt = job >> 1, nt = job & 1;
        const int pos = mt * 16 + m;
        const bool valid = pos < 49;
        const int oy = pos / 7, ox = pos - oy * 7;
        const _Float16* Bcol = ws + W2T_OFF + (nt * 16 + m) * 576;
        v8f acc = {};
#pragma unroll 1
        for (int kc = 0; kc < 18; ++kc) {
            const int kyx = kc >> 1, cb = (kc & 1) * 32;
            const int ky = kyx / 3, kx = kyx - ky * 3;
            const int pix = valid ? ((oy + ky) * 9 + ox + kx) : 81;  // zero row
            acc = wmma(load_a16(h1_s + pix * 64 + cb, lane),
                       load_b16(Bcol + kc * 32, lane), acc);
        }
        store_c(lane, acc, [&](int mm, int n, float v) {
            const int pos2 = mt * 16 + mm, oc = nt * 16 + n;
            const float r = LEAKY(v + bias_s[224 + oc]);
            h2_s[pos2 * 32 + oc] = (pos2 < 49) ? (_Float16)r : (_Float16)0;
        });
    }
    __syncthreads();

    // ---- phase 7: conv3 (32->8, 7x7->5x5); features -> ws[F_OFF] -----------
    _Float16* Frow = ws + F_OFF + (size_t)s * 224;
    for (int job = wave; job < 2; job += 4) {
        const int mt = job;
        const int pos = mt * 16 + m;
        const bool valid = pos < 25;
        const int oy = pos / 5, ox = pos - oy * 5;
        const _Float16* Bcol = ws + W3T_OFF + m * 288;  // rows oc>=8 zero
        v8f acc = {};
#pragma unroll 1
        for (int kc = 0; kc < 9; ++kc) {
            const int ky = kc / 3, kx = kc - ky * 3;
            const int pix = valid ? ((oy + ky) * 7 + ox + kx) : 49;  // zero row
            acc = wmma(load_a16(h2_s + pix * 32, lane),
                       load_b16(Bcol + kc * 32, lane), acc);
        }
        store_c(lane, acc, [&](int mm, int n, float v) {
            const int pos2 = mt * 16 + mm;
            if (pos2 < 25 && n < 8)  // NCHW flatten c*25+pos
                Frow[n * 25 + pos2] = (_Float16)LEAKY(v + bias_s[256 + n]);
        });
    }
    if (tid < 24) Frow[200 + tid] = (_Float16)0;  // K padding for fc1
}

// ---- batched FC kernel: 64 samples/block, whole chain as WMMA GEMMs --------

__global__ __launch_bounds__(128) void fc_chain(Params p, const _Float16* ws) {
    const int tid  = threadIdx.x;
    const int wave = tid >> 5;
    const int lane = tid & 31;
    const int m    = lane & 31 & 15;
    const int row0 = blockIdx.x * 64;

    __shared__ __align__(32) _Float16 a1[64 * 64];
    __shared__ __align__(32) _Float16 a2[64 * 32];
    __shared__ __align__(32) _Float16 a3[64 * 32];
    __shared__ __align__(32) _Float16 a4[64 * 32];
    __shared__ float bias_s[128];  // f1b@0(64) f2b@64(32) f3b@96(16) f4b@112(8) f5b@120(2)

    for (int i = tid; i < 122; i += 128) {
        float v;
        if (i < 64)       v = p.f1b[i];
        else if (i < 96)  v = p.f2b[i - 64];
        else if (i < 112) v = p.f3b[i - 96];
        else if (i < 120) v = p.f4b[i - 112];
        else              v = p.f5b[i - 120];
        bias_s[i] = v;
    }
    for (int i = tid; i < 64 * 32; i += 128) { a3[i] = (_Float16)0; a4[i] = (_Float16)0; }
    __syncthreads();

    const _Float16* F = ws + F_OFF + (size_t)row0 * 224;

    // fc1: [64x224]@[224x64], K chunks = 7
    for (int job = wave; job < 16; job += 4) {
        const int mt = job >> 2, nt = job & 3;
        const _Float16* Arow = F + (mt * 16 + m) * 224;
        const _Float16* Bcol = ws + F1T_OFF + (nt * 16 + m) * 224;
        v8f acc = {};
#pragma unroll 1
        for (int kc = 0; kc < 7; ++kc)
            acc = wmma(load_a16(Arow + kc * 32, lane), load_b16(Bcol + kc * 32, lane), acc);
        store_c(lane, acc, [&](int mm, int n, float v) {
            a1[(mt * 16 + mm) * 64 + nt * 16 + n] =
                (_Float16)LEAKY(v + bias_s[nt * 16 + n]);
        });
    }
    __syncthreads();

    // fc2: [64x64]@[64x32]
    for (int job = wave; job < 8; job += 4) {
        const int mt = job >> 1, nt = job & 1;
        const _Float16* Arow = a1 + (mt * 16 + m) * 64;
        const _Float16* Bcol = ws + F2T_OFF + (nt * 16 + m) * 64;
        v8f acc = {};
#pragma unroll
        for (int kc = 0; kc < 2; ++kc)
            acc = wmma(load_a16(Arow + kc * 32, lane), load_b16(Bcol + kc * 32, lane), acc);
        store_c(lane, acc, [&](int mm, int n, float v) {
            a2[(mt * 16 + mm) * 32 + nt * 16 + n] =
                (_Float16)LEAKY(v + bias_s[64 + nt * 16 + n]);
        });
    }
    __syncthreads();

    // fc3: [64x32]@[32x16] (a3 cols 16..31 stay zero for fc4 K-padding)
    {
        const int mt = wave;
        v8f acc = wmma(load_a16(a2 + (mt * 16 + m) * 32, lane),
                       load_b16(ws + F3T_OFF + m * 32, lane), v8f{});
        store_c(lane, acc, [&](int mm, int n, float v) {
            a3[(mt * 16 + mm) * 32 + n] = (_Float16)LEAKY(v + bias_s[96 + n]);
        });
    }
    __syncthreads();

    // fc4: [64x16pad32]@[16x8pad16]; cols n>=8 write 0 (K-padding for fc5)
    {
        const int mt = wave;
        v8f acc = wmma(load_a16(a3 + (mt * 16 + m) * 32, lane),
                       load_b16(ws + F4T_OFF + m * 32, lane), v8f{});
        store_c(lane, acc, [&](int mm, int n, float v) {
            const float r = LEAKY(v + bias_s[112 + n]);
            a4[(mt * 16 + mm) * 32 + n] = (n < 8) ? (_Float16)r : (_Float16)0;
        });
    }
    __syncthreads();

    // fc5: [64x8pad32]@[8x2pad16] -> d_out
    {
        const int mt = wave;
        v8f acc = wmma(load_a16(a4 + (mt * 16 + m) * 32, lane),
                       load_b16(ws + F5T_OFF + m * 32, lane), v8f{});
        store_c(lane, acc, [&](int mm, int n, float v) {
            if (n < 2)
                p.out[(size_t)(row0 + mt * 16 + mm) * 2 + n] = v + bias_s[120 + n];
        });
    }
}

extern "C" void kernel_launch(void* const* d_in, const int* in_sizes, int n_in,
                              void* d_out, int out_size, void* d_ws, size_t ws_size,
                              hipStream_t stream) {
    (void)n_in; (void)out_size; (void)ws_size;  // d_ws must hold ~7.2 MB
    Params p;
    p.x   = (const float*)d_in[0];
    p.c1w = (const float*)d_in[1];  p.c1b = (const float*)d_in[2];
    p.thw = (const float*)d_in[3];  p.thb = (const float*)d_in[4];
    p.phw = (const float*)d_in[5];  p.phb = (const float*)d_in[6];
    p.gw  = (const float*)d_in[7];  p.gb  = (const float*)d_in[8];
    p.ow  = (const float*)d_in[9];  p.ob  = (const float*)d_in[10];
    p.c2w = (const float*)d_in[11]; p.c2b = (const float*)d_in[12];
    p.c3w = (const float*)d_in[13]; p.c3b = (const float*)d_in[14];
    p.f1w = (const float*)d_in[15]; p.f1b = (const float*)d_in[16];
    p.f2w = (const float*)d_in[17]; p.f2b = (const float*)d_in[18];
    p.f3w = (const float*)d_in[19]; p.f3b = (const float*)d_in[20];
    p.f4w = (const float*)d_in[21]; p.f4b = (const float*)d_in[22];
    p.f5w = (const float*)d_in[23]; p.f5b = (const float*)d_in[24];
    p.out = (float*)d_out;

    _Float16* ws = (_Float16*)d_ws;
    const int B = in_sizes[0] / 121;

    prep_weights<<<64, 256, 0, stream>>>(p, ws);
    braggnn_fused<<<B, 128, 0, stream>>>(p, ws);
    fc_chain<<<B / 64, 128, 0, stream>>>(p, ws);
}
